// MYNDGraphTransformer_249108103246
// MI455X (gfx1250) — matrix operand order repaired and checked by reading, not compiled, not run on hardware
//
#include <hip/hip_runtime.h>
#include <hip/hip_bf16.h>
#include <math.h>

typedef __bf16 bf16_t;
typedef __attribute__((ext_vector_type(16))) __bf16 v16bf;
typedef __attribute__((ext_vector_type(8)))  float  v8f;

#define GN   2048   // tokens
#define GD   512    // model dim
#define GFF  2048
#define GH   8
#define GHD  64

// ---------------------------------------------------------------- helpers
static __device__ __forceinline__ v8f wmma_bf16(v16bf a, v16bf b, v8f c) {
  return __builtin_amdgcn_wmma_f32_16x16x32_bf16(false, a, false, b, (short)0, c,
                                                 false, false);
}

static __device__ __forceinline__ float gelu_exact(float x) {
  return 0.5f * x * (1.0f + erff(x * 0.70710678118654752f));
}

static __device__ __forceinline__ unsigned int pack_bf2(float lo, float hi) {
  unsigned short lb = __builtin_bit_cast(unsigned short, (__bf16)lo);
  unsigned short hb = __builtin_bit_cast(unsigned short, (__bf16)hi);
  return ((unsigned int)hb << 16) | (unsigned int)lb;
}

static __device__ __forceinline__ uint4 pack_bf8(const float4 f0, const float4 f1) {
  uint4 u;
  u.x = pack_bf2(f0.x, f0.y);
  u.y = pack_bf2(f0.z, f0.w);
  u.z = pack_bf2(f1.x, f1.y);
  u.w = pack_bf2(f1.z, f1.w);
  return u;
}

// A-fragment: 16x32 (MxK) bf16 row-major in LDS (one contiguous 32B run/lane).
static __device__ __forceinline__ v16bf load_a_frag(const bf16_t* p, int ld) {
  const int lane = threadIdx.x & 31;
  const int m  = lane & 15;
  const int kb = (lane >> 4) << 4;
  const bf16_t* row = p + m * ld + kb;
  v16bf a;
#pragma unroll
  for (int i = 0; i < 8; ++i) a[i] = row[i];
#pragma unroll
  for (int i = 0; i < 8; ++i) a[8 + i] = row[8 + i];
  return a;
}

// B-fragment from fragment-ordered LDS: [lane][16 halves] contiguous.
static __device__ __forceinline__ v16bf load_b_frag_lin(const bf16_t* p) {
  const int lane = threadIdx.x & 31;
  const bf16_t* q = p + lane * 16;
  v16bf b;
#pragma unroll
  for (int i = 0; i < 16; ++i) b[i] = q[i];
  return b;
}

// B-fragment from [n][k] row-major storage (K matrix keys x dims; B = K^T).
static __device__ __forceinline__ v16bf load_bT_frag(const bf16_t* p, int ld) {
  const int lane = threadIdx.x & 31;
  const int n  = lane & 15;
  const int kb = (lane >> 4) << 4;
  const bf16_t* row = p + n * ld + kb;
  v16bf b;
#pragma unroll
  for (int i = 0; i < 16; ++i) b[i] = row[i];
  return b;
}

// ---------------------------------------------------------------- GEMM kernel
// C[M,N] = act(A[M,K] @ B[K,N] + bias) [+ resid]; bf16 WMMA, f32 accumulate.
// 256 threads = 8 waves (4m x 2n), block tile 128x128, wave tile 32x64.
#define BM 128
#define BN 128
#define BK 32
#define ASTR 40

template <bool GELU, bool RESID>
__global__ __launch_bounds__(256) void gemm_bf16_wmma(
    const float* __restrict__ A, const float* __restrict__ B,
    const float* __restrict__ bias, const float* __restrict__ resid,
    float* __restrict__ C, int M, int N, int Kd) {
  __shared__ alignas(16) bf16_t As[BM * ASTR];
  __shared__ alignas(16) bf16_t Bs[(BN / 16) * 512];  // [subtile][lane][16]

  const int tid  = threadIdx.x;
  const int lane = tid & 31;
  const int w    = tid >> 5;
  const int wm   = w >> 1;
  const int wn   = w & 1;
  const int m0   = blockIdx.y * BM;
  const int n0   = blockIdx.x * BN;

  int ar[2], ac[2];
#pragma unroll
  for (int j = 0; j < 2; ++j) {
    int cidx = tid + 256 * j;
    ar[j] = cidx >> 2;
    ac[j] = (cidx & 3) * 8;
  }
  int br[8], bc[8], boff[8];
#pragma unroll
  for (int j = 0; j < 8; ++j) {
    int pidx = tid + 256 * j;
    int c = pidx & 127;
    int r = (pidx >> 7) * 2;
    br[j] = r;
    bc[j] = c;
    boff[j] = (c >> 4) * 512 + ((c & 15) + ((r >> 4) << 4)) * 16 + (r & 15);
  }

  v8f acc[2][4];
#pragma unroll
  for (int i = 0; i < 2; ++i)
#pragma unroll
    for (int j = 0; j < 4; ++j) acc[i][j] = (v8f){0, 0, 0, 0, 0, 0, 0, 0};

  float4 pa[2][2];
  float2 pb[8];
  auto fetch = [&](int k0) {
#pragma unroll
    for (int j = 0; j < 2; ++j) {
      const float* src = &A[(size_t)(m0 + ar[j]) * Kd + k0 + ac[j]];
      pa[j][0] = *(const float4*)src;
      pa[j][1] = *(const float4*)(src + 4);
    }
#pragma unroll
    for (int j = 0; j < 8; ++j) {
      pb[j].x = B[(size_t)(k0 + br[j]) * N + n0 + bc[j]];
      pb[j].y = B[(size_t)(k0 + br[j] + 1) * N + n0 + bc[j]];
    }
  };

  fetch(0);
#pragma unroll 1
  for (int k0 = 0; k0 < Kd; k0 += BK) {
#pragma unroll
    for (int j = 0; j < 2; ++j)
      *(uint4*)&As[ar[j] * ASTR + ac[j]] = pack_bf8(pa[j][0], pa[j][1]);
#pragma unroll
    for (int j = 0; j < 8; ++j)
      *(unsigned int*)&Bs[boff[j]] = pack_bf2(pb[j].x, pb[j].y);
    __syncthreads();

    if (k0 + BK < Kd) fetch(k0 + BK);   // overlap next global loads with WMMA

    v16bf af0 = load_a_frag(&As[(wm * 32 + 0) * ASTR], ASTR);
    v16bf af1 = load_a_frag(&As[(wm * 32 + 16) * ASTR], ASTR);
#pragma unroll
    for (int tn = 0; tn < 4; ++tn) {
      v16bf bfv = load_b_frag_lin(&Bs[(wn * 4 + tn) * 512]);
      acc[0][tn] = wmma_bf16(af0, bfv, acc[0][tn]);
      acc[1][tn] = wmma_bf16(af1, bfv, acc[1][tn]);
    }
    __syncthreads();
  }

  const int hf = lane >> 4;
  const int nl = lane & 15;
#pragma unroll
  for (int tm = 0; tm < 2; ++tm)
#pragma unroll
    for (int tn = 0; tn < 4; ++tn)
#pragma unroll
      for (int r = 0; r < 8; ++r) {
        int m = m0 + wm * 32 + tm * 16 + r + hf * 8;
        int n = n0 + wn * 64 + tn * 16 + nl;
        float v = acc[tm][tn][r] + bias[n];
        if (GELU)  v = gelu_exact(v);
        if (RESID) v += resid[(size_t)m * N + n];
        C[(size_t)m * N + n] = v;
      }
}

// ---------------------------------------------------------------- LayerNorm
__global__ __launch_bounds__(256) void ln_kernel(
    const float* __restrict__ x, const float* __restrict__ g,
    const float* __restrict__ b, float* __restrict__ out, int D, int act) {
  int row  = blockIdx.x * 8 + (threadIdx.x >> 5);
  int lane = threadIdx.x & 31;
  const float* xr = x + (size_t)row * D;
  float s = 0.f, s2 = 0.f;
  for (int j = lane; j < D; j += 32) { float v = xr[j]; s += v; s2 += v * v; }
#pragma unroll
  for (int o = 16; o; o >>= 1) {
    s  += __shfl_xor(s, o, 32);
    s2 += __shfl_xor(s2, o, 32);
  }
  float mean = s / D;
  float var  = s2 / D - mean * mean;
  float inv  = rsqrtf(var + 1e-5f);
  float* orow = out + (size_t)row * D;
  for (int j = lane; j < D; j += 32) {
    float v = (xr[j] - mean) * inv * g[j] + b[j];
    if (act) v = gelu_exact(v);
    orow[j] = v;
  }
}

// ---------------------------------------------------------------- pos-enc gather
__global__ __launch_bounds__(256) void posin_kernel(
    const int* __restrict__ depths, const int* __restrict__ degrees,
    const float* __restrict__ depth_emb, const float* __restrict__ degree_emb,
    const float* __restrict__ cent_W, const float* __restrict__ cent_b,
    float* __restrict__ out) {
  int n  = blockIdx.x;
  int di = min(max(depths[n], 0), 19);
  int gi = min(max(degrees[n], 0), 49);
  float* row = out + (size_t)n * 1536;
  for (int j = threadIdx.x; j < 1536; j += 256) {
    float v;
    if (j < 512)       v = depth_emb[di * 512 + j];
    else if (j < 1024) v = degree_emb[gi * 512 + (j - 512)];
    else               v = 0.5f * cent_W[j - 1024] + cent_b[j - 1024];
    row[j] = v;
  }
}

// ---------------------------------------------------------------- fused flash attention
#define FBR 16
#define FBC 32
#define KSTR 520

__global__ __launch_bounds__(256, 2) void flash_attn_kernel(
    const float* __restrict__ Q, const float* __restrict__ Kb,
    const float* __restrict__ Vb, const float* __restrict__ adj,
    const float* __restrict__ eW1, const float* __restrict__ eb1,
    const float* __restrict__ eW2, const float* __restrict__ eb2,
    const float* __restrict__ ebias, float* __restrict__ out) {
  __shared__ alignas(16) bf16_t Ks[FBC * KSTR];          // [key][dim] row-major
  __shared__ alignas(16) bf16_t Vs[32 * 512];            // frag order
  __shared__ alignas(16) bf16_t bias8[FBR * FBC * GH];   // [m][n][h]
  __shared__ alignas(16) bf16_t Ps[GH][FBR * 40];

  const int tid  = threadIdx.x;
  const int lane = tid & 31;
  const int h    = tid >> 5;
  const int r0   = blockIdx.x * FBR;

  v16bf qf[2];
  {
    const int m  = lane & 15;
    const int kb = (lane >> 4) << 4;
    const float* qrow = Q + (size_t)(r0 + m) * GD + h * GHD;
#pragma unroll
    for (int kk = 0; kk < 2; ++kk) {
#pragma unroll
      for (int i = 0; i < 8; ++i) qf[kk][i]     = (bf16_t)qrow[kk * 32 + kb + i];
#pragma unroll
      for (int i = 0; i < 8; ++i) qf[kk][8 + i] = (bf16_t)qrow[kk * 32 + kb + 8 + i];
    }
  }

  v8f o_acc[4];
#pragma unroll
  for (int t = 0; t < 4; ++t) o_acc[t] = (v8f){0, 0, 0, 0, 0, 0, 0, 0};
  float m_run[8], l_run[8];
#pragma unroll
  for (int r = 0; r < 8; ++r) { m_run[r] = -INFINITY; l_run[r] = 0.f; }

  const int hf = lane >> 4;
  const int nl = lane & 15;

#pragma unroll 1
  for (int c0 = 0; c0 < GN; c0 += FBC) {
    // K tile: row-major, vectorized
#pragma unroll
    for (int j = 0; j < 8; ++j) {
      int cidx = tid + 256 * j;
      int r  = cidx >> 6;
      int c8 = (cidx & 63) * 8;
      const float* src = &Kb[(size_t)(c0 + r) * GD + c8];
      float4 f0 = *(const float4*)src;
      float4 f1 = *(const float4*)(src + 4);
      *(uint4*)&Ks[r * KSTR + c8] = pack_bf8(f0, f1);
    }
    // V tile: fragment order, key-pairs packed
#pragma unroll
    for (int j = 0; j < 16; ++j) {
      int pidx = tid + 256 * j;
      int d2 = (pidx & 255) * 2;
      int r  = (pidx >> 8) * 2;
      const float2 va  = *(const float2*)&Vb[(size_t)(c0 + r) * GD + d2];
      const float2 vb2 = *(const float2*)&Vb[(size_t)(c0 + r + 1) * GD + d2];
      int base = (d2 >> 4) * 512 + ((d2 & 15) + ((r >> 4) << 4)) * 16 + (r & 15);
      *(unsigned int*)&Vs[base]      = pack_bf2(va.x, vb2.x);
      *(unsigned int*)&Vs[base + 16] = pack_bf2(va.y, vb2.y);
    }
    // cooperative edge-MLP bias for all 8 heads; one b128 store per element
#pragma unroll
    for (int it = 0; it < 2; ++it) {
      int idx = tid + 256 * it;
      int m = idx >> 5, n = idx & 31;
      float a = adj[(size_t)(r0 + m) * GN + (c0 + n)];
      float e[GH];
#pragma unroll
      for (int hh = 0; hh < GH; ++hh) e[hh] = eb2[hh] + a * ebias[hh];
#pragma unroll
      for (int j = 0; j < 16; ++j) {
        float t = fmaxf(a * eW1[j] + eb1[j], 0.f);
#pragma unroll
        for (int hh = 0; hh < GH; ++hh) e[hh] += t * eW2[j * GH + hh];
      }
      uint4 pk;
      pk.x = pack_bf2(e[0], e[1]);
      pk.y = pack_bf2(e[2], e[3]);
      pk.z = pack_bf2(e[4], e[5]);
      pk.w = pack_bf2(e[6], e[7]);
      *(uint4*)&bias8[idx * GH] = pk;
    }
    __syncthreads();

    // S = Q_h K_h^T : [16 x 32]
    v8f s[2];
    s[0] = (v8f){0, 0, 0, 0, 0, 0, 0, 0};
    s[1] = (v8f){0, 0, 0, 0, 0, 0, 0, 0};
#pragma unroll
    for (int nt = 0; nt < 2; ++nt)
#pragma unroll
      for (int kk = 0; kk < 2; ++kk) {
        v16bf bfv = load_bT_frag(&Ks[(nt * 16) * KSTR + h * GHD + kk * 32], KSTR);
        s[nt] = wmma_bf16(qf[kk], bfv, s[nt]);
      }

    // scale + edge bias; tile row-max
    float tmax[8];
#pragma unroll
    for (int r = 0; r < 8; ++r) tmax[r] = -INFINITY;
#pragma unroll
    for (int nt = 0; nt < 2; ++nt)
#pragma unroll
      for (int r = 0; r < 8; ++r) {
        int m = r + hf * 8;
        int n = nt * 16 + nl;
        float v = s[nt][r] * 0.125f + (float)bias8[(m * FBC + n) * GH + h];
        s[nt][r] = v;
        tmax[r] = fmaxf(tmax[r], v);
      }
#pragma unroll
    for (int o = 1; o < 16; o <<= 1)
#pragma unroll
      for (int r = 0; r < 8; ++r) tmax[r] = fmaxf(tmax[r], __shfl_xor(tmax[r], o, 32));

    // online softmax
    float corr[8], rsum[8];
#pragma unroll
    for (int r = 0; r < 8; ++r) {
      float mn = fmaxf(m_run[r], tmax[r]);
      corr[r] = __expf(m_run[r] - mn);
      m_run[r] = mn;
      rsum[r] = 0.f;
    }
#pragma unroll
    for (int nt = 0; nt < 2; ++nt)
#pragma unroll
      for (int r = 0; r < 8; ++r) {
        float p = __expf(s[nt][r] - m_run[r]);
        s[nt][r] = p;
        rsum[r] += p;
      }
#pragma unroll
    for (int o = 1; o < 16; o <<= 1)
#pragma unroll
      for (int r = 0; r < 8; ++r) rsum[r] += __shfl_xor(rsum[r], o, 32);
#pragma unroll
    for (int r = 0; r < 8; ++r) l_run[r] = l_run[r] * corr[r] + rsum[r];
#pragma unroll
    for (int t = 0; t < 4; ++t)
#pragma unroll
      for (int r = 0; r < 8; ++r) o_acc[t][r] *= corr[r];

    // P -> LDS -> A-fragment; O += P @ V_h
    bf16_t* myP = &Ps[h][0];
#pragma unroll
    for (int nt = 0; nt < 2; ++nt)
#pragma unroll
      for (int r = 0; r < 8; ++r) {
        int m = r + hf * 8;
        int n = nt * 16 + nl;
        myP[m * 40 + n] = (bf16_t)s[nt][r];
      }
    v16bf pf = load_a_frag(myP, 40);
#pragma unroll
    for (int t = 0; t < 4; ++t) {
      v16bf bfv = load_b_frag_lin(&Vs[(h * 4 + t) * 512]);
      o_acc[t] = wmma_bf16(pf, bfv, o_acc[t]);
    }
    __syncthreads();
  }

#pragma unroll
  for (int t = 0; t < 4; ++t)
#pragma unroll
    for (int r = 0; r < 8; ++r) {
      int m = r0 + r + hf * 8;
      int n = h * GHD + t * 16 + nl;
      out[(size_t)m * GD + n] = o_acc[t][r] / l_run[r];
    }
}

// ---------------------------------------------------------------- driver
static void gemm_call(const float* A, const float* B, const float* bias,
                      const float* resid, float* C, int M, int N, int Kd,
                      bool gelu, hipStream_t stream) {
  dim3 blk(256), grd(N / BN, M / BM);
  if (gelu)       gemm_bf16_wmma<true,  false><<<grd, blk, 0, stream>>>(A, B, bias, nullptr, C, M, N, Kd);
  else if (resid) gemm_bf16_wmma<false, true ><<<grd, blk, 0, stream>>>(A, B, bias, resid,  C, M, N, Kd);
  else            gemm_bf16_wmma<false, false><<<grd, blk, 0, stream>>>(A, B, bias, nullptr, C, M, N, Kd);
}

extern "C" void kernel_launch(void* const* d_in, const int* in_sizes, int n_in,
                              void* d_out, int out_size, void* d_ws, size_t ws_size,
                              hipStream_t stream) {
  (void)in_sizes; (void)n_in; (void)out_size; (void)ws_size;
  const float* node_emb   = (const float*)d_in[0];
  const float* adj        = (const float*)d_in[1];
  const int*   depths     = (const int*)d_in[2];
  const int*   degrees    = (const int*)d_in[3];
  const float* in_W       = (const float*)d_in[4];
  const float* in_b       = (const float*)d_in[5];
  const float* in_g       = (const float*)d_in[6];
  const float* in_beta    = (const float*)d_in[7];
  const float* depth_emb  = (const float*)d_in[8];
  const float* degree_emb = (const float*)d_in[9];
  const float* cent_W     = (const float*)d_in[10];
  const float* cent_b     = (const float*)d_in[11];
  const float* comb_W     = (const float*)d_in[12];
  const float* comb_b     = (const float*)d_in[13];
  const float* Wq   = (const float*)d_in[14];
  const float* bq   = (const float*)d_in[15];
  const float* Wk   = (const float*)d_in[16];
  const float* bk   = (const float*)d_in[17];
  const float* Wv   = (const float*)d_in[18];
  const float* bv   = (const float*)d_in[19];
  const float* Wo   = (const float*)d_in[20];
  const float* bo   = (const float*)d_in[21];
  const float* eW1  = (const float*)d_in[22];
  const float* eb1  = (const float*)d_in[23];
  const float* eW2  = (const float*)d_in[24];
  const float* eb2  = (const float*)d_in[25];
  const float* ebia = (const float*)d_in[26];
  const float* n1_g = (const float*)d_in[27];
  const float* n1_b = (const float*)d_in[28];
  const float* n2_g = (const float*)d_in[29];
  const float* n2_b = (const float*)d_in[30];
  const float* fW1  = (const float*)d_in[31];
  const float* fb1  = (const float*)d_in[32];
  const float* fW2  = (const float*)d_in[33];
  const float* fb2  = (const float*)d_in[34];
  const float* fin_g = (const float*)d_in[35];
  const float* fin_b = (const float*)d_in[36];

  const int N = GN, D = GD, FF = GFF, DIN = 384;
  const size_t ND = (size_t)N * D;
  float* X  = (float*)d_ws;
  float* X2 = X  + ND;
  float* NX = X2 + ND;
  float* Qb = NX + ND;
  float* Kb = Qb + ND;
  float* Vb = Kb + ND;
  float* AO = Vb + ND;
  float* Hb = AO + ND;   // N*FF floats; reused as [N,1536] pos-enc input

  dim3 blk(256);

  gemm_call(node_emb, in_W, in_b, nullptr, NX, N, D, DIN, false, stream);
  ln_kernel<<<N / 8, blk, 0, stream>>>(NX, in_g, in_beta, X, D, 1);

  posin_kernel<<<N, blk, 0, stream>>>(depths, degrees, depth_emb, degree_emb,
                                      cent_W, cent_b, Hb);
  gemm_call(Hb, comb_W, comb_b, X, X2, N, D, 1536, false, stream);

  float* cur = X2;
  float* alt = X;
  for (int i = 0; i < 3; ++i) {
    const size_t dd = (size_t)i * D * D;
    ln_kernel<<<N / 8, blk, 0, stream>>>(cur, n1_g + i * D, n1_b + i * D, NX, D, 0);
    gemm_call(NX, Wq + dd, bq + i * D, nullptr, Qb, N, D, D, false, stream);
    gemm_call(NX, Wk + dd, bk + i * D, nullptr, Kb, N, D, D, false, stream);
    gemm_call(NX, Wv + dd, bv + i * D, nullptr, Vb, N, D, D, false, stream);
    flash_attn_kernel<<<N / FBR, blk, 0, stream>>>(Qb, Kb, Vb, adj,
        eW1 + i * 16, eb1 + i * 16, eW2 + i * 16 * GH, eb2 + i * GH,
        ebia + i * GH, AO);
    gemm_call(AO, Wo + dd, bo + i * D, cur, alt, N, D, D, false, stream);
    { float* t = cur; cur = alt; alt = t; }
    ln_kernel<<<N / 8, blk, 0, stream>>>(cur, n2_g + i * D, n2_b + i * D, NX, D, 0);
    gemm_call(NX, fW1 + (size_t)i * D * FF, fb1 + i * FF, nullptr, Hb, N, FF, D, true, stream);
    gemm_call(Hb, fW2 + (size_t)i * FF * D, fb2 + i * D, cur, alt, N, D, FF, false, stream);
    { float* t = cur; cur = alt; alt = t; }
  }
  ln_kernel<<<N / 8, blk, 0, stream>>>(cur, fin_g, fin_b, (float*)d_out, D, 0);
}
